// MultiHeadAttention_64699387347677
// MI455X (gfx1250) — matrix operand order repaired and checked by reading, not compiled, not run on hardware
//
#include <hip/hip_runtime.h>

// ---------------------------------------------------------------------------
// MHA forward for MI455X (gfx1250): bf16 WMMA GEMMs + flash attention.
// B=2, S=2048, D_MODEL=1024, H=16, D_K=64.
// Round 3: register-pressure shaping (stay under 256 VGPRs, no s_set_vgpr_msb),
// hardware bf16 converts (v_cvt_pk_bf16_f32) instead of integer RNE sequences.
// ---------------------------------------------------------------------------

#define S_LEN 2048
#define DM    1024
#define NHEAD 16
#define DK    64

typedef __bf16 bf16_t;
typedef __attribute__((ext_vector_type(16))) __bf16 v16bf;
typedef __attribute__((ext_vector_type(4)))  __bf16 v4bf;
typedef __attribute__((ext_vector_type(8)))  float  v8f;

union FragU {
  v16bf bf;
  uint4 q[2];
};

// A-fragment (16x32, MxK, bf16), per CDNA5 ISA 7.12.2:
//   lane<16 : row=lane,    VGPR0-3 = K 0..7,  VGPR4-7 = K 16..23
//   lane>=16: row=lane-16, VGPR0-3 = K 8..15, VGPR4-7 = K 24..31
__device__ __forceinline__ v16bf load_a_frag(const bf16_t* base, int stride, int lane) {
  const int row = lane & 15;
  const int hi  = lane >> 4;
  const bf16_t* p = base + row * stride + hi * 8;
  FragU f;
  f.q[0] = *reinterpret_cast<const uint4*>(p);        // K = hi*8 .. +7
  f.q[1] = *reinterpret_cast<const uint4*>(p + 16);   // K = 16 + hi*8 .. +7
  return f.bf;
}

// B-fragment (32x16, KxN, bf16); memory layout: column n of B is a contiguous
// row n (stride `stride`) in memory (k contiguous).
//   lane<16 : col=lane,    K 0..7, 16..23 ; lane>=16: col=lane-16, K 8..15, 24..31
__device__ __forceinline__ v16bf load_b_frag(const bf16_t* base, int stride, int lane) {
  const int col = lane & 15;
  const int kh  = (lane >> 4) * 8;
  const bf16_t* p = base + col * stride + kh;
  FragU f;
  f.q[0] = *reinterpret_cast<const uint4*>(p);
  f.q[1] = *reinterpret_cast<const uint4*>(p + 16);
  return f.bf;
}

__device__ __forceinline__ v8f wmma_bf16(v16bf a, v16bf b, v8f c) {
  return __builtin_amdgcn_wmma_f32_16x16x32_bf16(
      /*neg_a=*/false, a, /*neg_b=*/false, b,
      /*c_mod=*/(short)0, c, /*reuse_a=*/false, /*reuse_b=*/false);
}

// ---------------------------------------------------------------------------
// fp32 -> bf16 conversion, 4 elements per thread (hardware cvt)
// ---------------------------------------------------------------------------
__global__ void cvt_bf16_kernel(const float* __restrict__ in,
                                bf16_t* __restrict__ out, int n4) {
  int i = blockIdx.x * blockDim.x + threadIdx.x;
  if (i < n4) {
    float4 f = reinterpret_cast<const float4*>(in)[i];
    v4bf o;
    o[0] = (bf16_t)f.x;
    o[1] = (bf16_t)f.y;
    o[2] = (bf16_t)f.z;
    o[3] = (bf16_t)f.w;
    reinterpret_cast<v4bf*>(out)[i] = o;
  }
}

// ---------------------------------------------------------------------------
// vt[bh][d][s] = vlin[(b*S + s)*DM + h*64 + d]
// ---------------------------------------------------------------------------
__global__ void transpose_v_kernel(const bf16_t* __restrict__ vlin,
                                   bf16_t* __restrict__ vt) {
  int idx = blockIdx.x * 256 + threadIdx.x;     // 2*16*64*2048 = 4,194,304
  int s  = idx & (S_LEN - 1);
  int d  = (idx >> 11) & (DK - 1);
  int bh = idx >> 17;
  int b = bh >> 4, h = bh & 15;
  vt[idx] = vlin[((size_t)(b * S_LEN + s)) * DM + h * DK + d];
}

// ---------------------------------------------------------------------------
// y[m,n] = sum_k X[m,k] * W[n,k] + bias[n]   (torch Linear: x @ W^T + b)
// Block 256 threads = 8 waves (2 in M x 4 in N), wave tile 64x64
// (4x4 WMMA accumulators), block tile 128x256.
// B-fragments streamed one at a time to keep live ranges short.
// ---------------------------------------------------------------------------
template <bool OUTF32>
__global__ void __launch_bounds__(256, 4)
gemm_bias_kernel(const bf16_t* __restrict__ X,
                 const bf16_t* __restrict__ W,
                 const float* __restrict__ bias,
                 void* __restrict__ out, int N, int K) {
  const int lane = threadIdx.x & 31;
  const int wave = threadIdx.x >> 5;
  const int wm = wave & 1, wn = wave >> 1;
  const int m0 = blockIdx.x * 128 + wm * 64;
  const int n0 = blockIdx.y * 256 + wn * 64;

  v8f acc[4][4];
#pragma unroll
  for (int i = 0; i < 4; ++i)
#pragma unroll
    for (int j = 0; j < 4; ++j)
#pragma unroll
      for (int r = 0; r < 8; ++r) acc[i][j][r] = 0.0f;

  for (int kk = 0; kk < K; kk += 32) {
    v16bf a[4];
#pragma unroll
    for (int i = 0; i < 4; ++i)
      a[i] = load_a_frag(X + (size_t)(m0 + 16 * i) * K + kk, K, lane);

    if (kk + 32 < K) {  // stream next K-chunk through WGP$ (global_prefetch_b8)
      __builtin_prefetch(X + (size_t)(m0 + (lane & 15)) * K + kk + 32, 0, 0);
      __builtin_prefetch(W + (size_t)(n0 + (lane & 15)) * K + kk + 32, 0, 0);
    }

#pragma unroll
    for (int j = 0; j < 4; ++j) {
      v16bf bj = load_b_frag(W + (size_t)(n0 + 16 * j) * K + kk, K, lane);
#pragma unroll
      for (int i = 0; i < 4; ++i)
        acc[i][j] = wmma_bf16(a[i], bj, acc[i][j]);
    }
  }

  const int c = lane & 15, hi = lane >> 4;
#pragma unroll
  for (int nj = 0; nj < 4; ++nj) {
    const int n = n0 + nj * 16 + c;
    const float bv = bias[n];
#pragma unroll
    for (int mi = 0; mi < 4; ++mi) {
#pragma unroll
      for (int r = 0; r < 8; ++r) {
        const int m = m0 + mi * 16 + r + 8 * hi;
        const float v = acc[mi][nj][r] + bv;
        if constexpr (OUTF32)
          reinterpret_cast<float*>(out)[(size_t)m * N + n] = v;
        else
          reinterpret_cast<bf16_t*>(out)[(size_t)m * N + n] = (bf16_t)v;
      }
    }
  }
}

// ---------------------------------------------------------------------------
// Causal flash attention. One wave per 32-row q group (two 16-row WMMA tiles
// sharing every K/V fragment load); 4 waves per block.
// qlin/klin: [B*S, DM] bf16 (head h at column h*64). vt: [B*H][64][S] bf16.
// sdpa out: [B*S, DM] bf16.
// ---------------------------------------------------------------------------
__global__ void __launch_bounds__(128, 4)
attention_kernel(const bf16_t* __restrict__ qlin,
                 const bf16_t* __restrict__ klin,
                 const bf16_t* __restrict__ vt,
                 bf16_t* __restrict__ sdpa) {
  __shared__ __align__(16) bf16_t plds[4][2][16 * 32];

  const int lane = threadIdx.x & 31;
  const int wave = threadIdx.x >> 5;
  const int bh = blockIdx.x;                 // 0..31
  const int b = bh >> 4, h = bh & 15;
  const int q0 = (blockIdx.y * 4 + wave) * 32;   // base of 32-row q group

  const bf16_t* qbase = qlin + (size_t)(b * S_LEN) * DM + h * DK;
  const bf16_t* kbase = klin + (size_t)(b * S_LEN) * DM + h * DK;
  const bf16_t* vbase = vt + (size_t)bh * DK * S_LEN;

  // q fragments: two 16-row tiles (u), d chunks 0..31 / 32..63 (dc)
  v16bf qa[2][2];
#pragma unroll
  for (int u = 0; u < 2; ++u)
#pragma unroll
    for (int dc = 0; dc < 2; ++dc)
      qa[u][dc] = load_a_frag(qbase + (size_t)(q0 + 16 * u) * DM + 32 * dc, DM, lane);

  float mrun[2][8], lrun[2][8];
  v8f accO[2][4];
#pragma unroll
  for (int u = 0; u < 2; ++u) {
#pragma unroll
    for (int r = 0; r < 8; ++r) { mrun[u][r] = -3.0e38f; lrun[u][r] = 0.0f; }
#pragma unroll
    for (int t = 0; t < 4; ++t)
#pragma unroll
      for (int r = 0; r < 8; ++r) accO[u][t][r] = 0.0f;
  }

  const int c = lane & 15;
  const int hi = lane >> 4;
  const float scale = 0.125f;                 // 1/sqrt(64)

  for (int kj0 = 0; kj0 <= q0 + 31; kj0 += 32) {
    // ---- scores: K fragments loaded per column-half, shared by both q tiles
    v8f s[2][2];
#pragma unroll
    for (int j = 0; j < 2; ++j) {
      v16bf kb0 = load_b_frag(kbase + (size_t)(kj0 + 16 * j) * DM, DM, lane);
      v16bf kb1 = load_b_frag(kbase + (size_t)(kj0 + 16 * j) * DM + 32, DM, lane);
#pragma unroll
      for (int u = 0; u < 2; ++u) {
        v8f z;
#pragma unroll
        for (int r = 0; r < 8; ++r) z[r] = 0.0f;
        z = wmma_bf16(qa[u][0], kb0, z);
        s[u][j] = wmma_bf16(qa[u][1], kb1, z);
      }
    }

    // ---- per-tile scale + causal mask + online softmax
    const int kc0 = kj0 + c;
    const int kc1 = kj0 + 16 + c;
    v16bf pa[2];
#pragma unroll
    for (int u = 0; u < 2; ++u) {
      float alpha[8];
#pragma unroll
      for (int r = 0; r < 8; ++r) {
        const int qpos = q0 + 16 * u + r + 8 * hi;
        float e0 = (kc0 <= qpos) ? s[u][0][r] * scale : -1.0e9f;
        float e1 = (kc1 <= qpos) ? s[u][1][r] * scale : -1.0e9f;
        float t = fmaxf(e0, e1);
        t = fmaxf(t, __shfl_xor(t, 1, 16));
        t = fmaxf(t, __shfl_xor(t, 2, 16));
        t = fmaxf(t, __shfl_xor(t, 4, 16));
        t = fmaxf(t, __shfl_xor(t, 8, 16));
        const float mnew = fmaxf(mrun[u][r], t);
        alpha[r] = __expf(mrun[u][r] - mnew);
        mrun[u][r] = mnew;
        const float p0 = __expf(e0 - mnew);
        const float p1 = __expf(e1 - mnew);
        s[u][0][r] = p0; s[u][1][r] = p1;
        float rs = p0 + p1;
        rs += __shfl_xor(rs, 1, 16);
        rs += __shfl_xor(rs, 2, 16);
        rs += __shfl_xor(rs, 4, 16);
        rs += __shfl_xor(rs, 8, 16);
        lrun[u][r] = lrun[u][r] * alpha[r] + rs;
      }
#pragma unroll
      for (int t = 0; t < 4; ++t)
#pragma unroll
        for (int r = 0; r < 8; ++r) accO[u][t][r] *= alpha[r];

      // P: C-layout fp32 -> LDS bf16 -> A-layout fragment (16x32).
      // DS ops from one wave are in-order (DScnt), so the load sees the stores.
      bf16_t* pw = plds[wave][u];
#pragma unroll
      for (int r = 0; r < 8; ++r) {
        const int m = r + 8 * hi;
        pw[m * 32 + c]      = (bf16_t)s[u][0][r];
        pw[m * 32 + 16 + c] = (bf16_t)s[u][1][r];
      }
      pa[u] = load_a_frag(pw, 32, lane);
    }

    // ---- O += P @ V : each V fragment feeds both q tiles (8 WMMA / 4 loads)
#pragma unroll
    for (int t = 0; t < 4; ++t) {
      v16bf vb = load_b_frag(vbase + (size_t)(t * 16) * S_LEN + kj0, S_LEN, lane);
      accO[0][t] = wmma_bf16(pa[0], vb, accO[0][t]);
      accO[1][t] = wmma_bf16(pa[1], vb, accO[1][t]);
    }
  }

  // ---- epilogue: divide by l, store bf16 into sdpa [B*S, DM] at head column
#pragma unroll
  for (int u = 0; u < 2; ++u) {
    bf16_t* obase = sdpa + (size_t)(b * S_LEN + q0 + 16 * u) * DM + h * DK;
#pragma unroll
    for (int r = 0; r < 8; ++r) {
      const float inv = 1.0f / lrun[u][r];
      const int m = r + 8 * hi;
#pragma unroll
      for (int t = 0; t < 4; ++t) {
        obase[(size_t)m * DM + t * 16 + c] = (bf16_t)(accO[u][t][r] * inv);
      }
    }
  }
}

// ---------------------------------------------------------------------------
// Host-side orchestration
// ---------------------------------------------------------------------------
extern "C" void kernel_launch(void* const* d_in, const int* in_sizes, int n_in,
                              void* d_out, int out_size, void* d_ws, size_t ws_size,
                              hipStream_t stream) {
  (void)in_sizes; (void)n_in; (void)out_size; (void)ws_size;

  const float* Q  = (const float*)d_in[0];
  const float* K  = (const float*)d_in[1];
  const float* V  = (const float*)d_in[2];
  // d_in[3] = causal mask (bool) -- computed analytically in-kernel
  const float* Wq = (const float*)d_in[4];
  const float* bq = (const float*)d_in[5];
  const float* Wk = (const float*)d_in[6];
  const float* bk = (const float*)d_in[7];
  const float* Wv = (const float*)d_in[8];
  const float* bv = (const float*)d_in[9];
  const float* Wo = (const float*)d_in[10];
  const float* bo = (const float*)d_in[11];
  float* out = (float*)d_out;

  const size_t EW  = (size_t)2 * S_LEN * DM;  // 4,194,304 activation elems
  const size_t EWW = (size_t)DM * DM;         // 1,048,576 weight elems

  bf16_t* p   = (bf16_t*)d_ws;
  bf16_t* Qb  = p; p += EW;
  bf16_t* Kb  = p; p += EW;
  bf16_t* Vb  = p; p += EW;
  bf16_t* Wqb = p; p += EWW;
  bf16_t* Wkb = p; p += EWW;
  bf16_t* Wvb = p; p += EWW;
  bf16_t* Wob = p; p += EWW;
  bf16_t* ql  = p; p += EW;
  bf16_t* kl  = p; p += EW;
  bf16_t* vl  = p; p += EW;
  bf16_t* vtv = Kb;  // alias: Kb dead after the K projection GEMM
  bf16_t* sd  = Qb;  // alias: Qb dead after the Q projection GEMM

  // 1) fp32 -> bf16 conversions
  cvt_bf16_kernel<<<(int)(EW / 4 / 256), 256, 0, stream>>>(Q, Qb, (int)(EW / 4));
  cvt_bf16_kernel<<<(int)(EW / 4 / 256), 256, 0, stream>>>(K, Kb, (int)(EW / 4));
  cvt_bf16_kernel<<<(int)(EW / 4 / 256), 256, 0, stream>>>(V, Vb, (int)(EW / 4));
  cvt_bf16_kernel<<<(int)(EWW / 4 / 256), 256, 0, stream>>>(Wq, Wqb, (int)(EWW / 4));
  cvt_bf16_kernel<<<(int)(EWW / 4 / 256), 256, 0, stream>>>(Wk, Wkb, (int)(EWW / 4));
  cvt_bf16_kernel<<<(int)(EWW / 4 / 256), 256, 0, stream>>>(Wv, Wvb, (int)(EWW / 4));
  cvt_bf16_kernel<<<(int)(EWW / 4 / 256), 256, 0, stream>>>(Wo, Wob, (int)(EWW / 4));

  // 2) Q/K/V projections (bf16 out): block tile 128x256, grid (32, 4)
  dim3 gg(2 * S_LEN / 128, DM / 256);
  gemm_bias_kernel<false><<<gg, 256, 0, stream>>>(Qb, Wqb, bq, ql, DM, DM);
  gemm_bias_kernel<false><<<gg, 256, 0, stream>>>(Kb, Wkb, bk, kl, DM, DM);
  gemm_bias_kernel<false><<<gg, 256, 0, stream>>>(Vb, Wvb, bv, vl, DM, DM);

  // 3) V transpose to [bh][d][s]
  transpose_v_kernel<<<(int)(EW / 256), 256, 0, stream>>>(vl, vtv);

  // 4) causal flash attention -> sdpa bf16 [B*S, DM]
  //    32 rows per wave, 4 waves per block: grid (B*H, S/32/4) = (32, 16)
  attention_kernel<<<dim3(2 * NHEAD, S_LEN / 32 / 4), 128, 0, stream>>>(ql, kl, vtv, sd);

  // 5) output projection (fp32 out)
  gemm_bias_kernel<true><<<gg, 256, 0, stream>>>(sd, Wob, bo, out, DM, DM);
}
